// ScopeProposalModel_28879360098635
// MI455X (gfx1250) — matrix operand order repaired and checked by reading, not compiled, not run on hardware
//
#include <hip/hip_runtime.h>

// ---------------------------------------------------------------------------
// ScopeProposalModel on MI455X (gfx1250, wave32, WMMA f16->f32)
// B=2, N=512, F_IN=64, H=128, L=3
// B-operands pre-packed into WMMA-fragment layout (one v16h per lane/frag);
// A-operands assembled from two aligned v8h runs (b128 loads + packed VALU).
// Edge-kernel weight staging uses GLOBAL_LOAD_ASYNC_TO_LDS_B128 (ASYNCcnt).
// ---------------------------------------------------------------------------

typedef __attribute__((ext_vector_type(16))) _Float16 v16h;
typedef __attribute__((ext_vector_type(8)))  _Float16 v8h;
typedef __attribute__((ext_vector_type(8)))  float    v8f;
typedef __attribute__((ext_vector_type(4)))  int      v4i;

#define BB   2
#define NN   512
#define FIN  64
#define HH   128
#define LL   3
#define NEGV (-1000000000.0f)

#ifndef __has_builtin
#define __has_builtin(x) 0
#endif
#if __has_builtin(__builtin_amdgcn_global_load_async_to_lds_b128) && \
    __has_builtin(__builtin_amdgcn_s_wait_asynccnt)
#define USE_ASYNC_LDS 1
#else
#define USE_ASYNC_LDS 0
#endif

// ---------------------------------------------------------------------------
// WMMA fragment layouts per CDNA5 ISA 7.12.2 (wave32):
// A (16x32 f16): lane L = row M=L&15; element e -> K = 16*(e>>3) + 8*(L>>4) + (e&7)
//   -> two contiguous 8-half runs at K = kh and K = 16+kh  (kh = 8*(L>>4))
// B (32x16 f16): lane L = col N=L&15; element e -> K = 16*(e>>3) + 8*(L>>4) + (e&7) (same as A)
// C/D (16x16 f32): VGPR r -> M = r + 8*(L>>4), N = L&15
// Packed-B: fragment f=(k0/32)*(N/16)+(n0/16); lane-major v16h per lane.
// ---------------------------------------------------------------------------

__device__ __forceinline__ v16h cat8(v8h lo, v8h hi) {
  return __builtin_shufflevector(lo, hi, 0, 1, 2, 3, 4, 5, 6, 7,
                                 8, 9, 10, 11, 12, 13, 14, 15);
}

__device__ __forceinline__ v16h load_a_frag(const _Float16* __restrict__ A,
                                            int lda, int k0, int lane) {
  const _Float16* p = A + (size_t)(lane & 15) * lda + k0 + ((lane >> 4) << 3);
  return cat8(*(const v8h*)p, *(const v8h*)(p + 16));
}

__device__ __forceinline__ v16h load_b_packed(const _Float16* __restrict__ Bp,
                                              int N, int k0, int col0, int lane) {
  const size_t frag = (size_t)(k0 >> 5) * (N >> 4) + (col0 >> 4);
  return *(const v16h*)(Bp + (frag * 32 + lane) * 16);
}

// ---------------------------------------------------------------------------
// Pack row-major f32 weight (K x N) -> packed f16 fragments (fused convert)
// ---------------------------------------------------------------------------
__global__ void pack_wfrag_kernel(const float* __restrict__ W,
                                  _Float16* __restrict__ out, int K, int N) {
  int idx = blockIdx.x * blockDim.x + threadIdx.x;
  const int total = K * N, stride = gridDim.x * blockDim.x;
  for (; idx < total; idx += stride) {
    const int k = idx / N, n = idx % N;
    const int frag = (k >> 5) * (N >> 4) + (n >> 4);
    const int lane = (((k >> 3) & 1) << 4) | (n & 15);
    out[((size_t)frag * 32 + lane) * 16 + ((((k >> 4) & 1) << 3) | (k & 7))] = (_Float16)W[idx];
  }
}

// Pack row-major f16 h (per batch, NN x HH) -> packed fragments (for adj@h)
__global__ void pack_hfrag_kernel(const _Float16* __restrict__ src,
                                  _Float16* __restrict__ dst) {
  const size_t base = (size_t)blockIdx.z * NN * HH;
  int idx = blockIdx.x * blockDim.x + threadIdx.x;
  const int stride = gridDim.x * blockDim.x;
  for (; idx < NN * HH; idx += stride) {
    const int k = idx / HH, n = idx % HH;
    const int frag = (k >> 5) * (HH >> 4) + (n >> 4);
    const int lane = (((k >> 3) & 1) << 4) | (n & 15);
    dst[base + ((size_t)frag * 32 + lane) * 16 + ((((k >> 4) & 1) << 3) | (k & 7))] = src[base + idx];
  }
}

// ---------------------------------------------------------------------------
// Generic dual-GEMM: C = act( A1@B1 [+ A2@B2] + bias1 + bias2 )
// One wave per 16x16 C tile; A row-major f16, B packed f16, f32 accum.
// ---------------------------------------------------------------------------
__global__ void __launch_bounds__(256)
wmma_gemm2_kernel(const _Float16* __restrict__ A1, long sA1, int lda1, int K1,
                  const _Float16* __restrict__ B1p, long sB1,
                  const _Float16* __restrict__ A2, long sA2, int lda2, int K2,
                  const _Float16* __restrict__ B2p, long sB2,
                  const float* __restrict__ bias1, const float* __restrict__ bias2,
                  float* __restrict__ C32, long sC32,
                  _Float16* __restrict__ C16, long sC16,
                  int M, int N, int relu) {
  const int lane = threadIdx.x & 31;
  const int wave = threadIdx.x >> 5;
  const int tiles_n = N >> 4;
  const int tiles   = (M >> 4) * tiles_n;
  const int tid = blockIdx.x * (blockDim.x >> 5) + wave;
  if (tid >= tiles) return;
  const int tm = tid / tiles_n, tn = tid % tiles_n;
  const int z = blockIdx.z;
  const int row0 = tm << 4, col0 = tn << 4;

  v8f acc = {};
  {
    const _Float16* A  = A1 + (size_t)z * sA1 + (size_t)row0 * lda1;
    const _Float16* Bp = B1p + (size_t)z * sB1;
    for (int k0 = 0; k0 < K1; k0 += 32) {
      v16h a = load_a_frag(A, lda1, k0, lane);
      v16h b = load_b_packed(Bp, N, k0, col0, lane);
      acc = __builtin_amdgcn_wmma_f32_16x16x32_f16(false, a, false, b,
                                                   (short)0, acc, false, false);
    }
  }
  if (A2) {
    const _Float16* A  = A2 + (size_t)z * sA2 + (size_t)row0 * lda2;
    const _Float16* Bp = B2p + (size_t)z * sB2;
    for (int k0 = 0; k0 < K2; k0 += 32) {
      v16h a = load_a_frag(A, lda2, k0, lane);
      v16h b = load_b_packed(Bp, N, k0, col0, lane);
      acc = __builtin_amdgcn_wmma_f32_16x16x32_f16(false, a, false, b,
                                                   (short)0, acc, false, false);
    }
  }

  const int col = col0 + (lane & 15);
  const float badd = (bias1 ? bias1[col] : 0.f) + (bias2 ? bias2[col] : 0.f);
#pragma unroll
  for (int r = 0; r < 8; ++r) {
    const int row = row0 + r + ((lane >> 4) << 3);
    float v = acc[r] + badd;
    if (relu) v = fmaxf(v, 0.f);
    if (C32) C32[(size_t)z * sC32 + (size_t)row * N + col] = v;
    if (C16) C16[(size_t)z * sC16 + (size_t)row * N + col] = (_Float16)v;
  }
}

// ---------------------------------------------------------------------------
// f32 -> f16 mirror (adjacency), grid-stride
// ---------------------------------------------------------------------------
__global__ void f32_to_f16_kernel(const float* __restrict__ src,
                                  _Float16* __restrict__ dst, int n) {
  int i = blockIdx.x * blockDim.x + threadIdx.x;
  const int stride = gridDim.x * blockDim.x;
  for (; i < n; i += stride) dst[i] = (_Float16)src[i];
}

// ---------------------------------------------------------------------------
// h0 = relu(X @ Win + b) -> f16    (K=64 only; VALU, negligible cost)
// ---------------------------------------------------------------------------
__global__ void __launch_bounds__(HH)
encode_kernel(const float* __restrict__ X, const float* __restrict__ W,
              const float* __restrict__ b, _Float16* __restrict__ h16) {
  const int node = blockIdx.x;  // B*N
  const int t = threadIdx.x;    // H
  const float* x = X + (size_t)node * FIN;
  float acc = b[t];
#pragma unroll 8
  for (int k = 0; k < FIN; ++k) acc += x[k] * W[k * HH + t];
  h16[(size_t)node * HH + t] = (_Float16)fmaxf(acc, 0.f);
}

// ---------------------------------------------------------------------------
// node scope head: (relu(h@sW1+sb1)@sW2+sb2) — f32, LDS tree reduction
// ---------------------------------------------------------------------------
__global__ void __launch_bounds__(HH)
scope_kernel(const float* __restrict__ hfin, const float* __restrict__ sW1,
             const float* __restrict__ sb1, const float* __restrict__ sW2,
             const float* __restrict__ sb2, float* __restrict__ out) {
  __shared__ float red[HH];
  const int node = blockIdx.x;
  const int t = threadIdx.x;
  const float* hrow = hfin + (size_t)node * HH;
  float acc = sb1[t];
#pragma unroll 8
  for (int k = 0; k < HH; ++k) acc += hrow[k] * sW1[k * HH + t];
  red[t] = fmaxf(acc, 0.f) * sW2[t];
  __syncthreads();
  for (int s = HH >> 1; s > 0; s >>= 1) {
    if (t < s) red[t] += red[t + s];
    __syncthreads();
  }
  if (t == 0) out[node] = red[0] + sb2[0];
}

// ---------------------------------------------------------------------------
// Edge MLP hot loop (~34 GF after P/Q factoring).
// hidden(i,j) = relu(P[i] + Q[j] + [|hi-hj|, hi*hj] @ Wcd + eb1)
// logit(i,j)  = hidden . eW2 + eb2
// Block = (i, 128 j's). Packed Wcd staged in LDS in two 32KB halves via
// GLOBAL_LOAD_ASYNC_TO_LDS_B128 (tracked by ASYNCcnt). A fragments built
// from two aligned v8h runs: ds_load_b128 (h_i) + global b128 (h_j) +
// packed v_pk f16 math, shufflevector'd into the v16h WMMA operand.
// ---------------------------------------------------------------------------
#define EK_WAVES 8
__global__ void __launch_bounds__(EK_WAVES * 32)
edge_kernel(const _Float16* __restrict__ h16,
            const float* __restrict__ P, const _Float16* __restrict__ Q16,
            const _Float16* __restrict__ Wcdp,   // packed: 64 frags (256x128)
            const float* __restrict__ eb1, const float* __restrict__ eW2,
            const float* __restrict__ eb2, float* __restrict__ Eraw) {
  const int b    = blockIdx.z;
  const int i    = blockIdx.x;
  const int lane = threadIdx.x & 31;
  const int wave = threadIdx.x >> 5;
  const int j0   = (blockIdx.y * EK_WAVES + wave) << 4;

  __shared__ __align__(32) _Float16 shW[32 * 512];  // 32 frags = 32KB
  __shared__ __align__(32) _Float16 sh_hi[HH];
  __shared__ float sh_pi[HH], sh_b1[HH], sh_w2[HH];
  if (threadIdx.x < HH) {
    const int t = threadIdx.x;
    sh_hi[t] = h16[((size_t)b * NN + i) * HH + t];
    sh_pi[t] = P[((size_t)b * NN + i) * HH + t];
    sh_b1[t] = eb1[t];
    sh_w2[t] = eW2[t];
  }

  const _Float16* hj = h16 + ((size_t)b * NN + (j0 + (lane & 15))) * HH;
  const int kh = (lane >> 4) << 3;

  v8f acc[8] = {};

  for (int half = 0; half < 2; ++half) {
    __syncthreads();
    // stage 32 packed fragments (32KB) of Wcd for this half into LDS
#if USE_ASYNC_LDS
    for (int c = threadIdx.x; c < 2048; c += EK_WAVES * 32) {  // 16B units
      __builtin_amdgcn_global_load_async_to_lds_b128(
          (v4i*)(Wcdp + (size_t)half * 16384 + (size_t)c * 8),
          (v4i*)(shW + (size_t)c * 8),
          0, 0);
    }
    __builtin_amdgcn_s_wait_asynccnt(0);
#else
    for (int c = threadIdx.x; c < 1024; c += EK_WAVES * 32) {  // 32B units
      *(v16h*)(shW + (size_t)c * 16) =
          *(const v16h*)(Wcdp + (size_t)half * 16384 + (size_t)c * 16);
    }
#endif
    if (half == 0)
      __builtin_prefetch(Wcdp + (size_t)16384, 0, 0);
    __syncthreads();

#pragma unroll
    for (int ks = 0; ks < 4; ++ks) {
      const int kb = ks << 5;  // feature-base within this half (0..96)
      // two aligned 8-half runs of the pair features
      const v8h hi0 = *(const v8h*)(sh_hi + kb + kh);
      const v8h hi1 = *(const v8h*)(sh_hi + kb + 16 + kh);
      const v8h hv0 = *(const v8h*)(hj + kb + kh);
      const v8h hv1 = *(const v8h*)(hj + kb + 16 + kh);
      v16h a;
      if (half == 0) {
        a = cat8(__builtin_elementwise_abs(hi0 - hv0),
                 __builtin_elementwise_abs(hi1 - hv1));
      } else {
        a = cat8(hi0 * hv0, hi1 * hv1);
      }
#pragma unroll
      for (int n = 0; n < 8; ++n) {
        v16h bf = *(const v16h*)(shW + ((size_t)((ks << 3) + n) * 32 + lane) * 16);
        acc[n] = __builtin_amdgcn_wmma_f32_16x16x32_f16(false, a, false, bf,
                                                        (short)0, acc[n], false, false);
      }
    }
  }

  // fused epilogue: +P[i] +Q[j] +eb1, relu, dot eW2, reduce over 16 lanes
  float partial[8];
#pragma unroll
  for (int r = 0; r < 8; ++r) partial[r] = 0.f;
#pragma unroll
  for (int n = 0; n < 8; ++n) {
    const int col = (n << 4) + (lane & 15);
    const float pi = sh_pi[col], b1 = sh_b1[col], w2 = sh_w2[col];
#pragma unroll
    for (int r = 0; r < 8; ++r) {
      const int j = j0 + r + ((lane >> 4) << 3);
      const float q = (float)Q16[((size_t)b * NN + j) * HH + col];
      float v = acc[n][r] + pi + q + b1;
      partial[r] += fmaxf(v, 0.f) * w2;
    }
  }
#pragma unroll
  for (int r = 0; r < 8; ++r) {
    float v = partial[r];
#pragma unroll
    for (int off = 8; off > 0; off >>= 1) v += __shfl_xor(v, off, 32);
    if ((lane & 15) == 0) {
      const int j = j0 + r + ((lane >> 4) << 3);
      Eraw[((size_t)b * NN + i) * NN + j] = v + eb2[0];
    }
  }
}

// ---------------------------------------------------------------------------
// edge_logits = 0.5*(E + E^T), diagonal = NEG
// ---------------------------------------------------------------------------
__global__ void sym_kernel(const float* __restrict__ Eraw, float* __restrict__ out) {
  size_t idx = (size_t)blockIdx.x * blockDim.x + threadIdx.x;
  const size_t total = (size_t)BB * NN * NN;
  if (idx >= total) return;
  const int j = (int)(idx % NN);
  const size_t t = idx / NN;
  const int i = (int)(t % NN);
  const int b = (int)(t / NN);
  out[idx] = (i == j) ? NEGV
                      : 0.5f * (Eraw[idx] + Eraw[((size_t)b * NN + j) * NN + i]);
}

// ---------------------------------------------------------------------------
extern "C" void kernel_launch(void* const* d_in, const int* in_sizes, int n_in,
                              void* d_out, int out_size, void* d_ws, size_t ws_size,
                              hipStream_t stream) {
  const float* node_feats = (const float*)d_in[0];
  const float* adj        = (const float*)d_in[1];
  const float* enc_in_W   = (const float*)d_in[2];
  const float* enc_in_b   = (const float*)d_in[3];
  const float* Wself      = (const float*)d_in[4];
  const float* bself      = (const float*)d_in[5];
  const float* Wmsg       = (const float*)d_in[6];
  const float* bmsg       = (const float*)d_in[7];
  const float* sW1        = (const float*)d_in[8];
  const float* sb1        = (const float*)d_in[9];
  const float* sW2        = (const float*)d_in[10];
  const float* sb2        = (const float*)d_in[11];
  const float* eW1        = (const float*)d_in[12];
  const float* eb1        = (const float*)d_in[13];
  const float* eW2        = (const float*)d_in[14];
  const float* eb2        = (const float*)d_in[15];

  // ---- outputs (tuple order, flat f32) ----
  float* out_lat   = (float*)d_out;                  // B*N*H
  float* out_scope = out_lat + (size_t)BB * NN * HH; // B*N
  float* out_edges = out_scope + (size_t)BB * NN;    // B*N*N

  // ---- workspace carve ----
  char* ws = (char*)d_ws;
  size_t off = 0;
  auto carve = [&](size_t bytes) -> char* {
    char* p = ws + off;
    off = (off + bytes + 255) & ~(size_t)255;
    return p;
  };
  _Float16* h16a     = (_Float16*)carve((size_t)BB * NN * HH * 2);
  _Float16* h16b     = (_Float16*)carve((size_t)BB * NN * HH * 2);
  _Float16* hpack    = (_Float16*)carve((size_t)BB * NN * HH * 2);
  _Float16* m16      = (_Float16*)carve((size_t)BB * NN * HH * 2);
  _Float16* adj16    = (_Float16*)carve((size_t)BB * NN * NN * 2);
  _Float16* Wselfpk  = (_Float16*)carve((size_t)LL * HH * HH * 2);
  _Float16* Wmsgpk   = (_Float16*)carve((size_t)LL * HH * HH * 2);
  _Float16* eW1pk    = (_Float16*)carve((size_t)4 * HH * HH * 2);
  float*    Pbuf     = (float*)carve((size_t)BB * NN * HH * 4);
  _Float16* Q16buf   = (_Float16*)carve((size_t)BB * NN * HH * 2);
  float*    Eraw     = (float*)carve((size_t)BB * NN * NN * 4);

  const long sH = (long)NN * HH;  // per-batch h stride

  // 1) f16 adjacency mirror + packed weights (fused f32->f16 + fragment pack)
  {
    int n = BB * NN * NN;
    int blocks = (n + 255) / 256; if (blocks > 2048) blocks = 2048;
    f32_to_f16_kernel<<<blocks, 256, 0, stream>>>(adj, adj16, n);
    for (int l = 0; l < LL; ++l) {
      pack_wfrag_kernel<<<(HH * HH + 255) / 256, 256, 0, stream>>>(
          Wself + (size_t)l * HH * HH, Wselfpk + (size_t)l * HH * HH, HH, HH);
      pack_wfrag_kernel<<<(HH * HH + 255) / 256, 256, 0, stream>>>(
          Wmsg + (size_t)l * HH * HH, Wmsgpk + (size_t)l * HH * HH, HH, HH);
    }
    pack_wfrag_kernel<<<(4 * HH * HH + 255) / 256, 256, 0, stream>>>(
        eW1, eW1pk, 4 * HH, HH);
  }

  // 2) encoder input projection
  encode_kernel<<<BB * NN, HH, 0, stream>>>(node_feats, enc_in_W, enc_in_b, h16a);

  // 3) GNN layers
  _Float16* hcur = h16a;
  _Float16* hnxt = h16b;
  const int gemm_tiles  = (NN / 16) * (HH / 16);  // 256
  const int gemm_blocks = (gemm_tiles + 7) / 8;   // 32
  for (int l = 0; l < LL; ++l) {
    // pack current h for use as a B-operand in adj@h
    pack_hfrag_kernel<<<dim3(64, 1, BB), 256, 0, stream>>>(hcur, hpack);
    // m = adj @ h   (K = 512)
    wmma_gemm2_kernel<<<dim3(gemm_blocks, 1, BB), 256, 0, stream>>>(
        adj16, (long)NN * NN, NN, NN, hpack, sH,
        nullptr, 0, 0, 0, nullptr, 0,
        nullptr, nullptr,
        nullptr, 0, m16, sH,
        NN, HH, /*relu=*/0);
    // h' = relu(h@Wself_l + m@Wmsg_l + bself_l + bmsg_l)
    wmma_gemm2_kernel<<<dim3(gemm_blocks, 1, BB), 256, 0, stream>>>(
        hcur, sH, HH, HH, Wselfpk + (size_t)l * HH * HH, 0,
        m16, sH, HH, HH, Wmsgpk + (size_t)l * HH * HH, 0,
        bself + (size_t)l * HH, bmsg + (size_t)l * HH,
        (l == LL - 1) ? out_lat : nullptr, sH, hnxt, sH,
        NN, HH, /*relu=*/1);
    _Float16* t = hcur; hcur = hnxt; hnxt = t;
  }

  // 4) node scope head (reads f32 latents written above)
  scope_kernel<<<BB * NN, HH, 0, stream>>>(out_lat, sW1, sb1, sW2, sb2, out_scope);

  // 5) P = h@eW1[0:H] (f32), Q = h@eW1[H:2H] (f16 mirror for the epilogue)
  wmma_gemm2_kernel<<<dim3(gemm_blocks, 1, BB), 256, 0, stream>>>(
      hcur, sH, HH, HH, eW1pk, 0,
      nullptr, 0, 0, 0, nullptr, 0,
      nullptr, nullptr, Pbuf, sH, nullptr, 0, NN, HH, 0);
  wmma_gemm2_kernel<<<dim3(gemm_blocks, 1, BB), 256, 0, stream>>>(
      hcur, sH, HH, HH, eW1pk + (size_t)32 * 512, 0,   // frags for k=128..255
      nullptr, 0, 0, 0, nullptr, 0,
      nullptr, nullptr, nullptr, 0, Q16buf, sH, NN, HH, 0);

  // 6) edge MLP hot loop (Wcd = packed frags for k=256..511 -> offset 64 frags)
  edge_kernel<<<dim3(NN, NN / (16 * EK_WAVES), BB), EK_WAVES * 32, 0, stream>>>(
      hcur, Pbuf, Q16buf, eW1pk + (size_t)64 * 512, eb1, eW2, eb2, Eraw);

  // 7) symmetrize + diagonal mask
  {
    const size_t total = (size_t)BB * NN * NN;
    sym_kernel<<<(int)((total + 255) / 256), 256, 0, stream>>>(Eraw, out_edges);
  }
}
